// WSDDNTransformerIMU_57947698757640
// MI455X (gfx1250) — compile-verified, matrix operand order checked by reading, not run on hardware
//
#include <hip/hip_runtime.h>
#include <math.h>

// ---------------- problem constants (from setup_inputs) ----------------
constexpr int kB   = 8;
constexpr int kC   = 512;
constexpr int kT   = 2048;
constexpr int kP   = 512;
constexpr int kD   = 512;
constexpr int kFF  = 1024;
constexpr int kNC  = 30;
constexpr int kNL  = 2;
constexpr int kNH  = 4;
constexpr int kDH  = 128;
constexpr int kSPP = 3584;           // C * (1+2+4)
constexpr int kM   = kB * kP;        // 4096 token rows

enum { FLAG_BIAS = 1, FLAG_ADD = 2, FLAG_GELU = 4, FLAG_RELU = 8 };

typedef __attribute__((ext_vector_type(16))) __bf16 v16bf;
typedef __attribute__((ext_vector_type(8)))  float  v8f;

// ---------------- small helpers ----------------
__device__ __forceinline__ unsigned short f2bf(float f) {
  unsigned int u = __float_as_uint(f);
  unsigned int r = u + 0x7FFFu + ((u >> 16) & 1u);   // round-to-nearest-even
  return (unsigned short)(r >> 16);
}

__device__ __forceinline__ v8f vzero() {
  v8f z;
#pragma unroll
  for (int i = 0; i < 8; ++i) z[i] = 0.f;
  return z;
}

__device__ __forceinline__ v8f wmma_bf16(v16bf a, v16bf b, v8f c) {
  // 8 args: (neg_a, A, neg_b, B, c_mod, C, reuse_a, reuse_b)
  return __builtin_amdgcn_wmma_f32_16x16x32_bf16(false, a, false, b, (short)0, c,
                                                 false, false);
}

// A fragment (16x32, 16-bit) from row-major [16 rows x K] at `src`, leading dim ld.
// ISA layout: lane m=l&15; VGPR i holds K pair k0 = 16*(i>=4) + 8*(l>=16) + 2*(i&3).
__device__ __forceinline__ v16bf load_frag_a(const unsigned short* src, int ld) {
  union { v16bf v; unsigned int u[8]; } f;
  int lane = threadIdx.x & 31;
  const unsigned short* r = src + (size_t)(lane & 15) * ld;
  int hb = (lane >> 4) << 3;
#pragma unroll
  for (int i = 0; i < 8; ++i) {
    int k0 = ((i & 4) << 2) + hb + ((i & 3) << 1);
    f.u[i] = *(const unsigned int*)(r + k0);
  }
  return f.v;
}

// B fragment when source is [K x N] row-major (not pre-transposed): lane holds
// column n=l&15, same K striping as A.
__device__ __forceinline__ v16bf load_frag_b_nt(const unsigned short* src, int ld) {
  union { v16bf v; unsigned short s[16]; } f;
  int lane = threadIdx.x & 31;
  int col  = lane & 15;
  int hb   = (lane >> 4) << 3;
#pragma unroll
  for (int i = 0; i < 8; ++i) {
    int k0 = ((i & 4) << 2) + hb + ((i & 3) << 1);
    f.s[2 * i]     = src[(size_t)k0 * ld + col];
    f.s[2 * i + 1] = src[(size_t)(k0 + 1) * ld + col];
  }
  return f.v;
}

// Core 32x64-per-wave WMMA tile loop. BNT=false: W is [N x K] (transposed
// weights, dot over K rows). BNT=true: W is [K x N] (e.g. attention V).
template <bool BNT>
__device__ __forceinline__ void gemm_tile_32x64(
    const unsigned short* __restrict__ A, int lda,
    const unsigned short* __restrict__ W, int ldw,
    int m0, int n0, int K, v8f acc[2][4]) {
  const int lr = threadIdx.x & 15;
  const int hs = (threadIdx.x & 16) ? 16 : 0;
  for (int k = 0; k < K; k += 32) {
    // pull the K-tile three steps ahead toward the WGP (global_prefetch_b8)
    if (k + 96 < K) {
      __builtin_prefetch(A + (size_t)(m0 + lr + hs) * lda + (k + 96), 0, 1);
      if (!BNT)
        __builtin_prefetch(W + (size_t)(n0 + lr + hs) * ldw + (k + 96), 0, 1);
      else
        __builtin_prefetch(W + (size_t)(k + 96 + lr) * ldw + n0, 0, 1);
    }
    v16bf a0 = load_frag_a(A + (size_t)m0 * lda + k, lda);
    v16bf a1 = load_frag_a(A + (size_t)(m0 + 16) * lda + k, lda);
    v16bf bf[4];
#pragma unroll
    for (int j = 0; j < 4; ++j) {
      if (BNT)
        bf[j] = load_frag_b_nt(W + (size_t)k * ldw + n0 + 16 * j, ldw);
      else
        bf[j] = load_frag_a(W + (size_t)(n0 + 16 * j) * ldw + k, ldw);
    }
#pragma unroll
    for (int j = 0; j < 4; ++j) {
      acc[0][j] = wmma_bf16(a0, bf[j], acc[0][j]);
      acc[1][j] = wmma_bf16(a1, bf[j], acc[1][j]);
    }
  }
}

// C/D layout: lane -> N = lane&15 ; VGPR v -> M = v + 8*(lane>=16)
__device__ __forceinline__ void store_tile(v8f acc, int m0, int n0,
                                           const float* bias,
                                           const float* add, int ldadd,
                                           float* outf, int ldo,
                                           unsigned short* outb, int ldob,
                                           int flags, float scale) {
  int lane = threadIdx.x & 31;
  int n  = n0 + (lane & 15);
  int mb = m0 + ((lane >> 4) << 3);
  float bv = (flags & FLAG_BIAS) ? bias[n] : 0.f;
#pragma unroll
  for (int v = 0; v < 8; ++v) {
    int m = mb + v;
    float x = acc[v] * scale + bv;
    if (flags & FLAG_ADD)  x += add[(size_t)m * ldadd + n];
    if (flags & FLAG_GELU) x = 0.5f * x * (1.f + erff(x * 0.7071067811865475f));
    if (flags & FLAG_RELU) x = fmaxf(x, 0.f);
    if (outf) outf[(size_t)m * ldo + n] = x;
    if (outb) outb[(size_t)m * ldob + n] = f2bf(x);
  }
}

// ---------------- generic WMMA GEMM: out = A[MxK] * W[NxK]^T (+bias,+add,act) ---
// grid = (N/128, M/64), block = 128 (4 waves, 2x2), wave tile 32x64.
__global__ __launch_bounds__(128) void gemm_bf16_kernel(
    const unsigned short* __restrict__ A, int lda,
    const unsigned short* __restrict__ W, int ldw,
    const float* __restrict__ bias,
    const float* __restrict__ addsrc, int ldadd,
    float* __restrict__ outf, int ldo,
    unsigned short* __restrict__ outb, int ldob,
    int K, int flags) {
  int wv = threadIdx.x >> 5;
  int m0 = blockIdx.y * 64 + (wv >> 1) * 32;
  int n0 = blockIdx.x * 128 + (wv & 1) * 64;
  v8f acc[2][4];
#pragma unroll
  for (int i = 0; i < 2; ++i)
#pragma unroll
    for (int j = 0; j < 4; ++j) acc[i][j] = vzero();

  gemm_tile_32x64<false>(A, lda, W, ldw, m0, n0, K, acc);

#pragma unroll
  for (int i = 0; i < 2; ++i)
#pragma unroll
    for (int j = 0; j < 4; ++j)
      store_tile(acc[i][j], m0 + i * 16, n0 + j * 16, bias, addsrc, ldadd,
                 outf, ldo, outb, ldob, flags, 1.0f);
}

// ---------------- attention: S = (Q K^T) * 1/sqrt(dh), batched over B*NH -------
// grid = (P/128, P/64, B*NH)
__global__ __launch_bounds__(128) void attn_scores_kernel(
    const unsigned short* __restrict__ qkv, float* __restrict__ S) {
  int z = blockIdx.z;                 // b*NH + h
  int b = z >> 2, h = z & 3;
  const int ld = 3 * kD;
  const unsigned short* Q  = qkv + (size_t)b * kP * ld + (size_t)h * kDH;
  const unsigned short* Km = Q + kD;
  float* Sout = S + (size_t)z * kP * kP;

  int wv = threadIdx.x >> 5;
  int m0 = blockIdx.y * 64 + (wv >> 1) * 32;
  int n0 = blockIdx.x * 128 + (wv & 1) * 64;
  v8f acc[2][4];
#pragma unroll
  for (int i = 0; i < 2; ++i)
#pragma unroll
    for (int j = 0; j < 4; ++j) acc[i][j] = vzero();

  gemm_tile_32x64<false>(Q, ld, Km, ld, m0, n0, kDH, acc);

  const float sc = 0.08838834764831845f;   // 1/sqrt(128)
#pragma unroll
  for (int i = 0; i < 2; ++i)
#pragma unroll
    for (int j = 0; j < 4; ++j)
      store_tile(acc[i][j], m0 + i * 16, n0 + j * 16, nullptr, nullptr, 0,
                 Sout, kP, nullptr, 0, 0, sc);
}

// ---------------- attention: O = softmax(S) @ V, batched ----------------------
// grid = (DH/128 = 1, P/64, B*NH)
__global__ __launch_bounds__(128) void attn_av_kernel(
    const unsigned short* __restrict__ Sb, const unsigned short* __restrict__ qkv,
    unsigned short* __restrict__ o) {
  int z = blockIdx.z;
  int b = z >> 2, h = z & 3;
  const int ld = 3 * kD;
  const unsigned short* A = Sb + (size_t)z * kP * kP;
  const unsigned short* V = qkv + (size_t)b * kP * ld + 2 * kD + (size_t)h * kDH;

  int wv = threadIdx.x >> 5;
  int m0 = blockIdx.y * 64 + (wv >> 1) * 32;
  int n0 = (wv & 1) * 64;                   // N = 128 covered by one block col
  v8f acc[2][4];
#pragma unroll
  for (int i = 0; i < 2; ++i)
#pragma unroll
    for (int j = 0; j < 4; ++j) acc[i][j] = vzero();

  gemm_tile_32x64<true>(A, kP, V, ld, m0, n0, kP, acc);

  unsigned short* ob = o + (size_t)(b * kP) * kD + (size_t)h * kDH;
#pragma unroll
  for (int i = 0; i < 2; ++i)
#pragma unroll
    for (int j = 0; j < 4; ++j)
      store_tile(acc[i][j], m0 + i * 16, n0 + j * 16, nullptr, nullptr, 0,
                 nullptr, 0, ob, kD, 0, 1.0f);
}

// ---------------- front-end kernels ------------------------------------------
__global__ __launch_bounds__(256) void f32_to_bf16_kernel(
    const float* __restrict__ in, unsigned short* __restrict__ out, int n) {
  int i = blockIdx.x * 256 + threadIdx.x;
  if (i < n) out[i] = f2bf(in[i]);
}

// inclusive prefix sum along T for each (b,c): csum[row][0..T]
__global__ __launch_bounds__(256) void prefix_scan_kernel(
    const float* __restrict__ x, float* __restrict__ csum) {
  int row = blockIdx.x;                              // b*C + c
  const float* src = x + (size_t)row * kT;
  float* dst = csum + (size_t)row * (kT + 1);
  __shared__ float sh[256];
  int tid = threadIdx.x;
  if (tid == 0) dst[0] = 0.f;
  float carry = 0.f;
  for (int base = 0; base < kT; base += 256) {
    sh[tid] = src[base + tid];
    __syncthreads();
    for (int off = 1; off < 256; off <<= 1) {
      float t = (tid >= off) ? sh[tid - off] : 0.f;
      __syncthreads();
      sh[tid] += t;
      __syncthreads();
    }
    dst[base + tid + 1] = sh[tid] + carry;
    float tot = sh[255];
    __syncthreads();
    carry += tot;
  }
}

__global__ void prep_kernel(const int* __restrict__ boxes, int* __restrict__ starts,
                            int* __restrict__ ends, float* __restrict__ centers) {
  int i = blockIdx.x * blockDim.x + threadIdx.x;
  if (i >= kB * kP) return;
  int s_raw = boxes[2 * i], e_raw = boxes[2 * i + 1];
  int s = min(max(s_raw, 0), kT - 1);
  int e = min(max(e_raw, 1), kT);
  if (e <= s) e = min(s + 1, kT);
  starts[i] = s;
  ends[i] = e;
  centers[i] = 0.5f * ((float)s_raw + (float)e_raw);
}

// bitonic sort of (center,index) per batch; tie-break on index == stable argsort
__global__ __launch_bounds__(512) void sort_kernel(
    const float* __restrict__ centers, int* __restrict__ sidx) {
  int b = blockIdx.x, tid = threadIdx.x;
  __shared__ float key[kP];
  __shared__ int   idx[kP];
  key[tid] = centers[b * kP + tid];
  idx[tid] = tid;
  __syncthreads();
  for (int k = 2; k <= kP; k <<= 1)
    for (int j = k >> 1; j > 0; j >>= 1) {
      int ixj = tid ^ j;
      if (ixj > tid) {
        float ka = key[tid], kb2 = key[ixj];
        int ia = idx[tid], ib = idx[ixj];
        bool agtb = (ka > kb2) || (ka == kb2 && ia > ib);
        bool up = ((tid & k) == 0);
        if (agtb == up) { key[tid] = kb2; key[ixj] = ka; idx[tid] = ib; idx[ixj] = ia; }
      }
      __syncthreads();
    }
  sidx[b * kP + tid] = idx[tid];
}

// exact SPP pooling into sorted order (bf16)
__global__ __launch_bounds__(256) void spp_kernel(
    const float* __restrict__ csum, const int* __restrict__ sidx,
    const int* __restrict__ starts, const int* __restrict__ ends,
    unsigned short* __restrict__ feats) {
  int rp = blockIdx.x;               // b*P + p_sorted
  int b = rp / kP;
  int orig = sidx[rp];
  int s = starts[b * kP + orig];
  int e = ends[b * kP + orig];
  int n = e - s;
  const float* cs = csum + (size_t)b * kC * (kT + 1);
  unsigned short* out = feats + (size_t)rp * kSPP;
  for (int t = threadIdx.x; t < kSPP; t += 256) {
    int c, i, L;
    if (t < kC)           { L = 1; c = t; i = 0; }
    else if (t < 3 * kC)  { L = 2; int u = t - kC;     c = u >> 1; i = u & 1; }
    else                  { L = 4; int u = t - 3 * kC; c = u >> 2; i = u & 3; }
    int lo = s + (i * n) / L;
    int hi = s + ((i + 1) * n + L - 1) / L;
    const float* row = cs + (size_t)c * (kT + 1);
    float mean = (row[hi] - row[lo]) / (float)(hi - lo);
    out[t] = f2bf(mean);
  }
}

// pos-MLP hidden layer (relu) in sorted order -> bf16 [4096,512]
__global__ __launch_bounds__(256) void pos_hidden_kernel(
    const int* __restrict__ boxes, const int* __restrict__ sidx,
    const float* __restrict__ w1, const float* __restrict__ b1,
    unsigned short* __restrict__ h) {
  int rp = blockIdx.x;
  int b = rp / kP;
  int orig = sidx[rp];
  int ix = b * kP + orig;
  float s = (float)boxes[2 * ix], e = (float)boxes[2 * ix + 1];
  float cen = 0.5f * (s + e);
  float len = fmaxf(e - s, 1.0f);
  float pf0 = cen / (float)kT, pf1 = len / (float)kT;
  for (int j = threadIdx.x; j < kD; j += 256) {
    float v = pf0 * w1[2 * j] + pf1 * w1[2 * j + 1] + b1[j];
    h[(size_t)rp * kD + j] = f2bf(fmaxf(v, 0.f));
  }
}

// ---------------- softmax / layernorm / epilogue kernels ----------------------
__global__ __launch_bounds__(256) void softmax_rows_kernel(
    const float* __restrict__ S, unsigned short* __restrict__ Sb) {
  int row = blockIdx.x;                      // (b*NH+h)*P + q
  const float* src = S + (size_t)row * kP;
  unsigned short* dst = Sb + (size_t)row * kP;
  __shared__ float sh[256];
  int tid = threadIdx.x;
  float v0 = src[tid], v1 = src[tid + 256];
  sh[tid] = fmaxf(v0, v1);
  __syncthreads();
  for (int off = 128; off > 0; off >>= 1) {
    if (tid < off) sh[tid] = fmaxf(sh[tid], sh[tid + off]);
    __syncthreads();
  }
  float mx = sh[0];
  __syncthreads();
  float e0 = expf(v0 - mx), e1 = expf(v1 - mx);
  sh[tid] = e0 + e1;
  __syncthreads();
  for (int off = 128; off > 0; off >>= 1) {
    if (tid < off) sh[tid] += sh[tid + off];
    __syncthreads();
  }
  float inv = 1.0f / sh[0];
  dst[tid] = f2bf(e0 * inv);
  dst[tid + 256] = f2bf(e1 * inv);
}

__global__ __launch_bounds__(256) void layernorm_kernel(
    const float* __restrict__ in, const float* __restrict__ g,
    const float* __restrict__ bta, float* __restrict__ outf,
    unsigned short* __restrict__ outb) {
  int row = blockIdx.x;
  const float* src = in + (size_t)row * kD;
  __shared__ float sh[256];
  int tid = threadIdx.x;
  float v0 = src[tid], v1 = src[tid + 256];
  sh[tid] = v0 + v1;
  __syncthreads();
  for (int off = 128; off > 0; off >>= 1) {
    if (tid < off) sh[tid] += sh[tid + off];
    __syncthreads();
  }
  float mean = sh[0] * (1.0f / kD);
  __syncthreads();
  float d0 = v0 - mean, d1 = v1 - mean;
  sh[tid] = d0 * d0 + d1 * d1;
  __syncthreads();
  for (int off = 128; off > 0; off >>= 1) {
    if (tid < off) sh[tid] += sh[tid + off];
    __syncthreads();
  }
  float rstd = rsqrtf(sh[0] * (1.0f / kD) + 1e-5f);
  float y0 = d0 * rstd * g[tid] + bta[tid];
  float y1 = d1 * rstd * g[tid + 256] + bta[tid + 256];
  outf[(size_t)row * kD + tid] = y0;
  outf[(size_t)row * kD + tid + 256] = y1;
  outb[(size_t)row * kD + tid] = f2bf(y0);
  outb[(size_t)row * kD + tid + 256] = f2bf(y1);
}

// scatter sorted tokens back to original proposal order: ctx[sidx[p]] = tok[p]
__global__ __launch_bounds__(256) void scatter_ctx_kernel(
    const float* __restrict__ tok, const int* __restrict__ sidx,
    float* __restrict__ ctx) {
  int rp = blockIdx.x;
  int b = rp / kP;
  int orig = sidx[rp];
  const float* src = tok + (size_t)rp * kD;
  float* dst = ctx + ((size_t)(b * kP + orig)) * kD;
  for (int i = threadIdx.x; i < kD; i += 256) dst[i] = src[i];
}

__global__ __launch_bounds__(64) void heads_kernel(
    const float* __restrict__ ctx, const float* __restrict__ cw,
    const float* __restrict__ cb, const float* __restrict__ dw,
    const float* __restrict__ db, float* __restrict__ cls,
    float* __restrict__ det) {
  int row = blockIdx.x;
  __shared__ float sh[kD];
  for (int i = threadIdx.x; i < kD; i += 64) sh[i] = ctx[(size_t)row * kD + i];
  __syncthreads();
  for (int o = threadIdx.x; o < 2 * kNC; o += 64) {
    int c = o % kNC;
    bool isdet = o >= kNC;
    const float* wrow = (isdet ? dw : cw) + (size_t)c * kD;
    float acc = isdet ? db[c] : cb[c];
    for (int k = 0; k < kD; ++k) acc += sh[k] * wrow[k];
    (isdet ? det : cls)[(size_t)row * kNC + c] = acc;
  }
}

__global__ __launch_bounds__(256) void cls_softmax_kernel(
    const float* __restrict__ logits, float* __restrict__ prob) {
  int row = blockIdx.x * 256 + threadIdx.x;
  if (row >= kM) return;
  const float* src = logits + (size_t)row * kNC;
  float mx = -1e30f;
  for (int c = 0; c < kNC; ++c) mx = fmaxf(mx, src[c]);
  float s = 0.f;
  for (int c = 0; c < kNC; ++c) s += expf(src[c] - mx);
  float inv = 1.0f / s;
  float* dst = prob + (size_t)row * kNC;
  for (int c = 0; c < kNC; ++c) dst[c] = expf(src[c] - mx) * inv;
}

__global__ __launch_bounds__(256) void det_softmax_kernel(
    const float* __restrict__ logits, float* __restrict__ prob) {
  int bc = blockIdx.x;                       // b*NC + c
  int b = bc / kNC, c = bc % kNC;
  const float* base = logits + (size_t)b * kP * kNC + c;
  float* pbase = prob + (size_t)b * kP * kNC + c;
  __shared__ float sh[256];
  int tid = threadIdx.x;
  float v0 = base[(size_t)tid * kNC], v1 = base[(size_t)(tid + 256) * kNC];
  sh[tid] = fmaxf(v0, v1);
  __syncthreads();
  for (int off = 128; off > 0; off >>= 1) {
    if (tid < off) sh[tid] = fmaxf(sh[tid], sh[tid + off]);
    __syncthreads();
  }
  float mx = sh[0];
  __syncthreads();
  float e0 = expf(v0 - mx), e1 = expf(v1 - mx);
  sh[tid] = e0 + e1;
  __syncthreads();
  for (int off = 128; off > 0; off >>= 1) {
    if (tid < off) sh[tid] += sh[tid + off];
    __syncthreads();
  }
  float inv = 1.0f / sh[0];
  pbase[(size_t)tid * kNC] = e0 * inv;
  pbase[(size_t)(tid + 256) * kNC] = e1 * inv;
}

__global__ __launch_bounds__(256) void joint_video_kernel(
    const float* __restrict__ cp, const float* __restrict__ dp,
    float* __restrict__ joint, float* __restrict__ video) {
  int bc = blockIdx.x;
  int b = bc / kNC, c = bc % kNC;
  int tid = threadIdx.x;
  float acc = 0.f;
  for (int p = tid; p < kP; p += 256) {
    size_t ix = ((size_t)b * kP + p) * kNC + c;
    float j = cp[ix] * dp[ix];
    joint[ix] = j;
    acc += j;
  }
  __shared__ float sh[256];
  sh[tid] = acc;
  __syncthreads();
  for (int off = 128; off > 0; off >>= 1) {
    if (tid < off) sh[tid] += sh[tid + off];
    __syncthreads();
  }
  if (tid == 0) video[b * kNC + c] = sh[0];
}

// ---------------- host-side orchestration -------------------------------------
extern "C" void kernel_launch(void* const* d_in, const int* in_sizes, int n_in,
                              void* d_out, int out_size, void* d_ws, size_t ws_size,
                              hipStream_t stream) {
  (void)in_sizes; (void)n_in; (void)out_size; (void)ws_size;
  const float* x      = (const float*)d_in[0];
  const int*   boxes  = (const int*)d_in[1];
  const float* proj_w = (const float*)d_in[2];
  const float* proj_b = (const float*)d_in[3];
  const float* pos_w1 = (const float*)d_in[4];
  const float* pos_b1 = (const float*)d_in[5];
  const float* pos_w2 = (const float*)d_in[6];
  const float* pos_b2 = (const float*)d_in[7];
  const float* qkv_w  = (const float*)d_in[8];
  const float* qkv_b  = (const float*)d_in[9];
  const float* out_w  = (const float*)d_in[10];
  const float* out_b  = (const float*)d_in[11];
  const float* ln1_g  = (const float*)d_in[12];
  const float* ln1_b  = (const float*)d_in[13];
  const float* ff1_w  = (const float*)d_in[14];
  const float* ff1_b  = (const float*)d_in[15];
  const float* ff2_w  = (const float*)d_in[16];
  const float* ff2_b  = (const float*)d_in[17];
  const float* ln2_g  = (const float*)d_in[18];
  const float* ln2_b  = (const float*)d_in[19];
  const float* cls_w  = (const float*)d_in[20];
  const float* cls_b  = (const float*)d_in[21];
  const float* det_w  = (const float*)d_in[22];
  const float* det_b  = (const float*)d_in[23];

  // d_out layout: video[240], joint[B*P*NC], class_logits, det_logits, ctx
  float* out   = (float*)d_out;
  float* video = out;
  float* joint = out + kB * kNC;
  float* clsl  = joint + (size_t)kM * kNC;
  float* detl  = clsl + (size_t)kM * kNC;
  float* ctx   = detl + (size_t)kM * kNC;

  // workspace carve-out
  char* wp = (char*)d_ws;
  auto alloc = [&](size_t bytes) -> void* {
    void* p = (void*)wp;
    wp += (bytes + 255) & ~(size_t)255;
    return p;
  };
  float* csum            = (float*)alloc(sizeof(float) * (size_t)kB * kC * (kT + 1));
  int*   starts          = (int*)alloc(4 * (size_t)kM);
  int*   ends            = (int*)alloc(4 * (size_t)kM);
  float* centers         = (float*)alloc(4 * (size_t)kM);
  int*   sidx            = (int*)alloc(4 * (size_t)kM);
  unsigned short* feats  = (unsigned short*)alloc(2ull * kM * kSPP);
  unsigned short* hpos   = (unsigned short*)alloc(2ull * kM * kD);
  unsigned short* proj_wb= (unsigned short*)alloc(2ull * kD * kSPP);
  unsigned short* pos_w2b= (unsigned short*)alloc(2ull * kD * kD);
  unsigned short* qkv_wb = (unsigned short*)alloc(2ull * kNL * 3 * kD * kD);
  unsigned short* out_wb = (unsigned short*)alloc(2ull * kNL * kD * kD);
  unsigned short* ff1_wb = (unsigned short*)alloc(2ull * kNL * kFF * kD);
  unsigned short* ff2_wb = (unsigned short*)alloc(2ull * kNL * kD * kFF);
  float* posemb          = (float*)alloc(4ull * kM * kD);
  float* tok             = (float*)alloc(4ull * kM * kD);
  unsigned short* tokb   = (unsigned short*)alloc(2ull * kM * kD);
  unsigned short* qkvb   = (unsigned short*)alloc(2ull * kM * 3 * kD);
  float* S               = (float*)alloc(4ull * kB * kNH * kP * kP);
  unsigned short* Sb     = (unsigned short*)alloc(2ull * kB * kNH * kP * kP);
  unsigned short* ob     = (unsigned short*)alloc(2ull * kM * kD);
  float* ybuf            = (float*)alloc(4ull * kM * kD);
  float* x1              = (float*)alloc(4ull * kM * kD);
  unsigned short* x1b    = (unsigned short*)alloc(2ull * kM * kD);
  unsigned short* h1b    = (unsigned short*)alloc(2ull * kM * kFF);
  float* cprob           = (float*)alloc(4ull * kM * kNC);
  float* dprob           = (float*)alloc(4ull * kM * kNC);

  // weights -> bf16
  auto cv = [&](const float* s, unsigned short* d, size_t n) {
    f32_to_bf16_kernel<<<dim3((unsigned)((n + 255) / 256)), dim3(256), 0, stream>>>(
        s, d, (int)n);
  };
  cv(proj_w, proj_wb, (size_t)kD * kSPP);
  cv(pos_w2, pos_w2b, (size_t)kD * kD);
  cv(qkv_w,  qkv_wb,  (size_t)kNL * 3 * kD * kD);
  cv(out_w,  out_wb,  (size_t)kNL * kD * kD);
  cv(ff1_w,  ff1_wb,  (size_t)kNL * kFF * kD);
  cv(ff2_w,  ff2_wb,  (size_t)kNL * kD * kFF);

  // front-end
  prefix_scan_kernel<<<kB * kC, 256, 0, stream>>>(x, csum);
  prep_kernel<<<(kM + 255) / 256, 256, 0, stream>>>(boxes, starts, ends, centers);
  sort_kernel<<<kB, kP, 0, stream>>>(centers, sidx);
  spp_kernel<<<kM, 256, 0, stream>>>(csum, sidx, starts, ends, feats);
  pos_hidden_kernel<<<kM, 256, 0, stream>>>(boxes, sidx, pos_w1, pos_b1, hpos);

  // pos_emb_s = relu(pf@W1^T+b1) @ W2^T + b2
  gemm_bf16_kernel<<<dim3(kD / 128, kM / 64), 128, 0, stream>>>(
      hpos, kD, pos_w2b, kD, pos_b2, nullptr, 0, posemb, kD, nullptr, 0, kD,
      FLAG_BIAS);
  // tok = feats_s @ proj_w^T + proj_b + pos_emb_s
  gemm_bf16_kernel<<<dim3(kD / 128, kM / 64), 128, 0, stream>>>(
      feats, kSPP, proj_wb, kSPP, proj_b, posemb, kD, tok, kD, tokb, kD, kSPP,
      FLAG_BIAS | FLAG_ADD);

  for (int l = 0; l < kNL; ++l) {
    const unsigned short* qw = qkv_wb + (size_t)l * 3 * kD * kD;
    const unsigned short* ow = out_wb + (size_t)l * kD * kD;
    const unsigned short* f1 = ff1_wb + (size_t)l * kFF * kD;
    const unsigned short* f2 = ff2_wb + (size_t)l * kD * kFF;

    // qkv = tok @ qkv_w^T + qkv_b  (bf16 only)
    gemm_bf16_kernel<<<dim3(3 * kD / 128, kM / 64), 128, 0, stream>>>(
        tokb, kD, qw, kD, qkv_b + (size_t)l * 3 * kD, nullptr, 0,
        nullptr, 0, qkvb, 3 * kD, kD, FLAG_BIAS);
    // S = Q K^T / sqrt(dh), batched over B*NH
    attn_scores_kernel<<<dim3(kP / 128, kP / 64, kB * kNH), 128, 0, stream>>>(qkvb, S);
    softmax_rows_kernel<<<kB * kNH * kP, 256, 0, stream>>>(S, Sb);
    // O = softmax(S) @ V  -> concat-head layout [4096, 512] bf16
    attn_av_kernel<<<dim3(1, kP / 64, kB * kNH), 128, 0, stream>>>(Sb, qkvb, ob);
    // y = tok + O @ out_w^T + out_b
    gemm_bf16_kernel<<<dim3(kD / 128, kM / 64), 128, 0, stream>>>(
        ob, kD, ow, kD, out_b + (size_t)l * kD, tok, kD, ybuf, kD, nullptr, 0,
        kD, FLAG_BIAS | FLAG_ADD);
    layernorm_kernel<<<kM, 256, 0, stream>>>(ybuf, ln1_g + (size_t)l * kD,
                                             ln1_b + (size_t)l * kD, x1, x1b);
    // h1 = gelu(x1 @ ff1^T + b) (bf16 only)
    gemm_bf16_kernel<<<dim3(kFF / 128, kM / 64), 128, 0, stream>>>(
        x1b, kD, f1, kD, ff1_b + (size_t)l * kFF, nullptr, 0, nullptr, 0, h1b,
        kFF, kD, FLAG_BIAS | FLAG_GELU);
    // y2 = x1 + h1 @ ff2^T + b
    gemm_bf16_kernel<<<dim3(kD / 128, kM / 64), 128, 0, stream>>>(
        h1b, kFF, f2, kFF, ff2_b + (size_t)l * kD, x1, kD, ybuf, kD, nullptr, 0,
        kFF, FLAG_BIAS | FLAG_ADD);
    layernorm_kernel<<<kM, 256, 0, stream>>>(ybuf, ln2_g + (size_t)l * kD,
                                             ln2_b + (size_t)l * kD, tok, tokb);
  }

  // back to original order + heads + probabilities
  scatter_ctx_kernel<<<kM, 256, 0, stream>>>(tok, sidx, ctx);
  heads_kernel<<<kM, 64, 0, stream>>>(ctx, cls_w, cls_b, det_w, det_b, clsl, detl);
  cls_softmax_kernel<<<(kM + 255) / 256, 256, 0, stream>>>(clsl, cprob);
  det_softmax_kernel<<<kB * kNC, 256, 0, stream>>>(detl, dprob);
  joint_video_kernel<<<kB * kNC, 256, 0, stream>>>(cprob, dprob, joint, video);
}
